// WanAVJointCrossAttention_69157563400758
// MI455X (gfx1250) — compile-verified, matrix operand order checked by reading, not run on hardware
//
#include <hip/hip_runtime.h>
#include <hip/hip_bf16.h>

#define T_N   21
#define H_N   30
#define W_N   52
#define HW_N  1560
#define C_N   1536
#define NH_N  12
#define D_N   128
#define M_N   32760
#define L2_N  162
#define S_N   16

typedef __bf16 bf16_t;
typedef bf16_t v16bf __attribute__((ext_vector_type(16)));
typedef float  v8f   __attribute__((ext_vector_type(8)));

union Frag { v16bf v; unsigned int u[8]; };

__device__ __forceinline__ unsigned short f2bf(float f) {
  union { float f; unsigned int u; } x; x.f = f;
  unsigned int u = x.u;
  return (unsigned short)((u + 0x7FFFu + ((u >> 16) & 1u)) >> 16);
}
__device__ __forceinline__ unsigned int f2bf_pk(float lo, float hi) {
  return (unsigned int)f2bf(lo) | ((unsigned int)f2bf(hi) << 16);
}
__device__ __forceinline__ float bf2f(unsigned int h) {
  union { unsigned int u; float f; } x; x.u = (h & 0xFFFFu) << 16;
  return x.f;
}

// ---------------------------------------------------------------- utility
__global__ __launch_bounds__(256) void zero_f32(float* p, int n) {
  const int i = blockIdx.x * 256 + threadIdx.x;
  if (i < n) p[i] = 0.0f;
}

// convert Wq and Wo (f32, [n][k] row-major) -> bf16
__global__ __launch_bounds__(256) void cvt_w(const float* __restrict__ a,
                                             const float* __restrict__ b,
                                             unsigned short* __restrict__ oa,
                                             unsigned short* __restrict__ ob) {
  const int i = (blockIdx.x * 256 + threadIdx.x) * 4;
  if (i < C_N * C_N) {
    float4 x = *(const float4*)(a + i);
    uint2 u; u.x = f2bf_pk(x.x, x.y); u.y = f2bf_pk(x.z, x.w);
    *(uint2*)(oa + i) = u;
    float4 y = *(const float4*)(b + i);
    uint2 v; v.x = f2bf_pk(y.x, y.y); v.y = f2bf_pk(y.z, y.w);
    *(uint2*)(ob + i) = v;
  }
}

// ---------------------------------------------------------------- heat gate (+ qh -> bf16 conversion)
__global__ __launch_bounds__(256) void heat_kernel(const float* __restrict__ qh,
                                                   const float* __restrict__ te,
                                                   const float* __restrict__ mod,
                                                   const float* __restrict__ Wh,
                                                   const float* __restrict__ bh,
                                                   float* __restrict__ heat,
                                                   unsigned short* __restrict__ qhb) {
  __shared__ float xs[C_N];
  __shared__ float red[256];
  __shared__ float sA, sB;
  const int r = blockIdx.x;
  const int tid = threadIdx.x;
  const float* row = qh + (size_t)r * C_N;
  float ps = 0.0f;
#pragma unroll
  for (int j = 0; j < 6; ++j) { const int c = tid + j * 256; const float v = row[c]; xs[c] = v; ps += v; }
  red[tid] = ps; __syncthreads();
  for (int st = 128; st > 0; st >>= 1) { if (tid < st) red[tid] += red[tid + st]; __syncthreads(); }
  if (tid == 0) sA = red[0] * (1.0f / C_N);
  __syncthreads();
  // fused: store bf16 copy of q_hidden (fits L2 -> GEMM-Q A re-reads stay on-chip)
  {
    const size_t rb = (size_t)r * C_N;
#pragma unroll
    for (int j = 0; j < 3; ++j) {
      const int p = tid + j * 256;  // 768 pairs
      *(unsigned int*)&qhb[rb + 2 * p] = f2bf_pk(xs[2 * p], xs[2 * p + 1]);
    }
  }
  const float mu = sA;
  ps = 0.0f;
#pragma unroll
  for (int j = 0; j < 6; ++j) { const int c = tid + j * 256; const float d = xs[c] - mu; ps += d * d; }
  red[tid] = ps; __syncthreads();
  for (int st = 128; st > 0; st >>= 1) { if (tid < st) red[tid] += red[tid + st]; __syncthreads(); }
  if (tid == 0) sB = rsqrtf(red[0] * (1.0f / C_N) + 1e-6f);
  __syncthreads();
  const float rv = sB;
  const float* terow = te + (size_t)r * C_N;
  ps = 0.0f;
#pragma unroll
  for (int j = 0; j < 6; ++j) {
    const int c = tid + j * 256;
    const float tev = terow[c];
    const float e0 = mod[c] + tev;
    const float e1 = mod[C_N + c] + tev;
    const float z = (xs[c] - mu) * rv;
    ps += (z * (1.0f + e1) + e0) * Wh[c];
  }
  red[tid] = ps; __syncthreads();
  for (int st = 128; st > 0; st >>= 1) { if (tid < st) red[tid] += red[tid + st]; __syncthreads(); }
  if (tid == 0) heat[r] = 1.0f / (1.0f + __expf(-(red[0] + bh[0])));
}

// ---------------------------------------------------------------- WMMA GEMM
// 64 (M) x 128 (N) tile per workgroup, Ktile = 64 (2 wmma k-steps per sync), 8 waves.
// MODE 1: epilogue = +bq, row-sumsq atomics, *gq, RoPE, bf16 store (q path).
// MODE 0: epilogue = +bo, f32 store to output (o path).
struct StageT { unsigned short a[64][66]; unsigned short b[128][66]; };
struct EpiT   { float e[64][130]; float psum[64][4]; };
union  ShU    { StageT st; EpiT ep; };

template <int MODE>
__global__ __launch_bounds__(256) void gemm_wmma(const unsigned short* __restrict__ Abf,
                                                 const unsigned short* __restrict__ Wbf,
                                                 const float* __restrict__ bias,
                                                 const float* __restrict__ gq,
                                                 const float* __restrict__ qfreqs,
                                                 float* __restrict__ rowsq,
                                                 unsigned short* __restrict__ qout,
                                                 float* __restrict__ fout) {
  __shared__ ShU sh;
  const int m0 = blockIdx.x * 64;
  const int n0 = blockIdx.y * 128;
  const int tid = threadIdx.x;
  const int lane = tid & 31;
  const int wv = tid >> 5;
  const int wm = wv & 1, wn = wv >> 1;
  const int hi = lane >> 4, mn = lane & 15;

  v8f acc00 = {}, acc01 = {}, acc10 = {}, acc11 = {};

  const int ar = tid >> 2, akq = (tid & 3) * 16;  // A staging: 64 rows x 64 k, 16 elems/thread
  int arow = m0 + ar; if (arow > M_N - 1) arow = M_N - 1;
  const int bn = tid >> 1, bkq = (tid & 1) * 32;  // B staging: 128 rows x 64 k, 32 elems/thread

  for (int k0 = 0; k0 < C_N; k0 += 64) {
    __syncthreads();
    {
      const unsigned short* ap = Abf + (size_t)arow * C_N + k0 + akq;
      const uint4 x = *(const uint4*)ap;
      const uint4 y = *(const uint4*)(ap + 8);
      unsigned int* dst = (unsigned int*)&sh.st.a[ar][akq];
      dst[0] = x.x; dst[1] = x.y; dst[2] = x.z; dst[3] = x.w;
      dst[4] = y.x; dst[5] = y.y; dst[6] = y.z; dst[7] = y.w;
    }
    {
      const unsigned short* bp = Wbf + (size_t)(n0 + bn) * C_N + k0 + bkq;
      const uint4 x0 = *(const uint4*)bp;
      const uint4 x1 = *(const uint4*)(bp + 8);
      const uint4 x2 = *(const uint4*)(bp + 16);
      const uint4 x3 = *(const uint4*)(bp + 24);
      unsigned int* dst = (unsigned int*)&sh.st.b[bn][bkq];
      dst[0]  = x0.x; dst[1]  = x0.y; dst[2]  = x0.z; dst[3]  = x0.w;
      dst[4]  = x1.x; dst[5]  = x1.y; dst[6]  = x1.z; dst[7]  = x1.w;
      dst[8]  = x2.x; dst[9]  = x2.y; dst[10] = x2.z; dst[11] = x2.w;
      dst[12] = x3.x; dst[13] = x3.y; dst[14] = x3.z; dst[15] = x3.w;
    }
    __syncthreads();

    // fragment gather per CDNA5 16-bit A/B layout:
    // lane m = lane&15, half = lane>>4; VGPR v holds K pair starting at
    // k = (v<4 ? 2v : 16+2(v-4)) + 8*half (+ ks)  -> 32-bit LDS loads.
#pragma unroll
    for (int ks = 0; ks < 64; ks += 32) {
      Frag a0, a1, b0, b1;
#pragma unroll
      for (int v = 0; v < 8; ++v) {
        const int kk = ((v < 4) ? 2 * v : 16 + 2 * (v - 4)) + 8 * hi + ks;
        a0.u[v] = *(const unsigned int*)&sh.st.a[wm * 32 + mn][kk];
        a1.u[v] = *(const unsigned int*)&sh.st.a[wm * 32 + 16 + mn][kk];
        b0.u[v] = *(const unsigned int*)&sh.st.b[wn * 32 + mn][kk];
        b1.u[v] = *(const unsigned int*)&sh.st.b[wn * 32 + 16 + mn][kk];
      }
      acc00 = __builtin_amdgcn_wmma_f32_16x16x32_bf16(false, a0.v, false, b0.v, (short)0, acc00, false, false);
      acc01 = __builtin_amdgcn_wmma_f32_16x16x32_bf16(false, a0.v, false, b1.v, (short)0, acc01, false, false);
      acc10 = __builtin_amdgcn_wmma_f32_16x16x32_bf16(false, a1.v, false, b0.v, (short)0, acc10, false, false);
      acc11 = __builtin_amdgcn_wmma_f32_16x16x32_bf16(false, a1.v, false, b1.v, (short)0, acc11, false, false);
    }
  }
  __syncthreads();

  // C/D layout: lane n = lane&15, VGPR r holds row r + 8*(lane>>4)
#pragma unroll
  for (int r8 = 0; r8 < 8; ++r8) {
    const int rr = wm * 32 + r8 + 8 * hi;
    const int cc = wn * 32 + mn;
    sh.ep.e[rr][cc]           = acc00[r8];
    sh.ep.e[rr][cc + 16]      = acc01[r8];
    sh.ep.e[rr + 16][cc]      = acc10[r8];
    sh.ep.e[rr + 16][cc + 16] = acc11[r8];
  }
  __syncthreads();

  if (MODE == 0) {
    for (int idx = tid; idx < 64 * 128; idx += 256) {
      const int r = idx >> 7, c = idx & 127;
      const int rg = m0 + r;
      if (rg < M_N) fout[(size_t)rg * C_N + n0 + c] = sh.ep.e[r][c] + bias[n0 + c];
    }
  } else {
    const int tpr = tid & 3, r = tid >> 2;
    const int rg = m0 + r;
    const int cb = tpr * 32;
    float ss = 0.0f;
#pragma unroll
    for (int c = 0; c < 32; ++c) {
      const float y = sh.ep.e[r][cb + c] + bias[n0 + cb + c];
      sh.ep.e[r][cb + c] = y;
      ss += y * y;
    }
    sh.ep.psum[r][tpr] = ss;
    __syncthreads();
    if (tpr == 0 && rg < M_N) {
      atomicAdd(&rowsq[rg],
                sh.ep.psum[r][0] + sh.ep.psum[r][1] + sh.ep.psum[r][2] + sh.ep.psum[r][3]);
    }
    if (rg < M_N) {
      const int pos = rg % HW_N;
      const int hh = pos / W_N, ww = pos % W_N;
#pragma unroll
      for (int p = 0; p < 16; ++p) {
        const int d = tpr * 16 + p;   // pair index within the head (col tile == head)
        const int c0 = 2 * d;
        const float xr = sh.ep.e[r][c0]     * gq[n0 + c0];
        const float xi = sh.ep.e[r][c0 + 1] * gq[n0 + c0 + 1];
        const int pidx = (d < 22) ? 0 : ((d < 43) ? hh : ww);
        const float fr = qfreqs[pidx * 128 + 2 * d];
        const float fi = qfreqs[pidx * 128 + 2 * d + 1];
        *(unsigned int*)&qout[(size_t)rg * C_N + n0 + c0] =
            f2bf_pk(xr * fr - xi * fi, xr * fi + xi * fr);
      }
    }
  }
}

// ---------------------------------------------------------------- K/V (336 rows)
__global__ __launch_bounds__(256) void kv_kernel(const float* __restrict__ kvh,
                                                 const float* __restrict__ Wk,
                                                 const float* __restrict__ bk,
                                                 const float* __restrict__ Wv,
                                                 const float* __restrict__ bv,
                                                 const float* __restrict__ gk,
                                                 const float* __restrict__ kvf,
                                                 float* __restrict__ kws,
                                                 float* __restrict__ vws) {
  __shared__ float xrow[C_N];
  __shared__ float proj[C_N];
  __shared__ float red[256];
  __shared__ float rnS;
  const int b = blockIdx.x;
  const int t = b / S_N, s = b % S_N;
  const int tid = threadIdx.x;
  int src = (t == 0 ? -11 : 8 * t - 10) + s;       // audio window index (analytic)
  src = src < 0 ? 0 : (src > L2_N - 1 ? L2_N - 1 : src);
#pragma unroll
  for (int j = 0; j < 6; ++j) { const int c = tid + j * 256; xrow[c] = kvh[(size_t)src * C_N + c]; }
  __syncthreads();
#pragma unroll
  for (int j = 0; j < 6; ++j) {
    const int n = tid + j * 256;
    const float* wr = Wk + (size_t)n * C_N;
    float acc = bk[n];
    for (int k = 0; k < C_N; k += 4) {
      const float4 w4 = *(const float4*)(wr + k);
      acc += xrow[k] * w4.x + xrow[k + 1] * w4.y + xrow[k + 2] * w4.z + xrow[k + 3] * w4.w;
    }
    proj[n] = acc;
  }
  __syncthreads();
  float ss = 0.0f;
#pragma unroll
  for (int j = 0; j < 6; ++j) { const int n = tid + j * 256; ss += proj[n] * proj[n]; }
  red[tid] = ss; __syncthreads();
  for (int st = 128; st > 0; st >>= 1) { if (tid < st) red[tid] += red[tid + st]; __syncthreads(); }
  if (tid == 0) rnS = rsqrtf(red[0] * (1.0f / C_N) + 1e-6f);
  __syncthreads();
  const float rn = rnS;
  const size_t ob = (size_t)b * C_N;
#pragma unroll
  for (int j = 0; j < 3; ++j) {                    // 768 RoPE pairs
    const int pp = tid + j * 256;
    const int nh = pp >> 6, d = pp & 63;
    const int c0 = nh * D_N + 2 * d;
    const float xr = proj[c0]     * rn * gk[c0];
    const float xi = proj[c0 + 1] * rn * gk[c0 + 1];
    const int pidx = (d < 22) ? s : 0;
    const float fr = kvf[pidx * 128 + 2 * d];
    const float fi = kvf[pidx * 128 + 2 * d + 1];
    kws[ob + c0]     = xr * fr - xi * fi;
    kws[ob + c0 + 1] = xr * fi + xi * fr;
  }
#pragma unroll
  for (int j = 0; j < 6; ++j) {
    const int n = tid + j * 256;
    const float* wr = Wv + (size_t)n * C_N;
    float acc = bv[n];
    for (int k = 0; k < C_N; k += 4) {
      const float4 w4 = *(const float4*)(wr + k);
      acc += xrow[k] * w4.x + xrow[k + 1] * w4.y + xrow[k + 2] * w4.z + xrow[k + 3] * w4.w;
    }
    vws[ob + n] = acc;
  }
}

// ---------------------------------------------------------------- attention
// one block per (head, t); K/V staged in LDS; q rows over threads.
// Applies deferred RMS scale (rowsq), softmax over 16, heat, writes o (bf16)
// in place over the q buffer (each (row,head) chunk owned by exactly one thread).
__global__ __launch_bounds__(256) void attn_kernel(const unsigned short* qbf,
                                                   const float* __restrict__ kws,
                                                   const float* __restrict__ vws,
                                                   const float* __restrict__ heat,
                                                   const float* __restrict__ rowsq,
                                                   unsigned short* obf) {
  const int head = blockIdx.x;
  const int t = blockIdx.y;
  const int tid = threadIdx.x;
  __shared__ float kls[S_N][D_N];
  __shared__ float vls[S_N][D_N];
  for (int idx = tid; idx < S_N * D_N; idx += 256) {
    const int s = idx >> 7, c = idx & 127;
    const size_t g = (size_t)(t * S_N + s) * C_N + head * D_N + c;
    kls[s][c] = kws[g];
    vls[s][c] = vws[g];
  }
  __syncthreads();
  const float scale = 0.08838834764831845f;  // 1/sqrt(128)
  for (int row = tid; row < HW_N; row += 256) {
    const int qrow = t * HW_N + row;
    const unsigned short* qp = qbf + (size_t)qrow * C_N + head * D_N;
    float sc[S_N];
#pragma unroll
    for (int s = 0; s < S_N; ++s) sc[s] = 0.0f;
    for (int cb = 0; cb < D_N; cb += 8) {
      const uint4 pk = *(const uint4*)(qp + cb);
      float q8[8];
      q8[0] = bf2f(pk.x); q8[1] = bf2f(pk.x >> 16);
      q8[2] = bf2f(pk.y); q8[3] = bf2f(pk.y >> 16);
      q8[4] = bf2f(pk.z); q8[5] = bf2f(pk.z >> 16);
      q8[6] = bf2f(pk.w); q8[7] = bf2f(pk.w >> 16);
#pragma unroll
      for (int s = 0; s < S_N; ++s) {
        float a = 0.0f;
#pragma unroll
        for (int e = 0; e < 8; ++e) a += q8[e] * kls[s][cb + e];
        sc[s] += a;
      }
    }
    const float rn = rsqrtf(rowsq[qrow] * (1.0f / C_N) + 1e-6f);
    const float ms = rn * scale;
    float mx = -1e30f;
#pragma unroll
    for (int s = 0; s < S_N; ++s) { sc[s] *= ms; mx = fmaxf(mx, sc[s]); }
    float sum = 0.0f;
#pragma unroll
    for (int s = 0; s < S_N; ++s) { sc[s] = __expf(sc[s] - mx); sum += sc[s]; }
    const float hs = heat[qrow] / sum;
    unsigned short* op = obf + (size_t)qrow * C_N + head * D_N;
    for (int cb = 0; cb < D_N; cb += 8) {
      float o8[8];
#pragma unroll
      for (int e = 0; e < 8; ++e) o8[e] = 0.0f;
#pragma unroll
      for (int s = 0; s < S_N; ++s) {
        const float p = sc[s];
#pragma unroll
        for (int e = 0; e < 8; ++e) o8[e] += p * vls[s][cb + e];
      }
      uint4 pk;
      pk.x = f2bf_pk(o8[0] * hs, o8[1] * hs);
      pk.y = f2bf_pk(o8[2] * hs, o8[3] * hs);
      pk.z = f2bf_pk(o8[4] * hs, o8[5] * hs);
      pk.w = f2bf_pk(o8[6] * hs, o8[7] * hs);
      *(uint4*)(op + cb) = pk;
    }
  }
}

// ---------------------------------------------------------------- launch
extern "C" void kernel_launch(void* const* d_in, const int* in_sizes, int n_in,
                              void* d_out, int out_size, void* d_ws, size_t ws_size,
                              hipStream_t stream) {
  (void)in_sizes; (void)n_in; (void)out_size; (void)ws_size;
  const float* qh   = (const float*)d_in[0];
  const float* kvh  = (const float*)d_in[1];
  const float* te   = (const float*)d_in[2];
  /* d_in[3] grid_sizes unused (constants baked in) */
  const float* qf   = (const float*)d_in[4];
  const float* kvf  = (const float*)d_in[5];
  const float* Wq   = (const float*)d_in[6];
  const float* bq   = (const float*)d_in[7];
  const float* Wk   = (const float*)d_in[8];
  const float* bk   = (const float*)d_in[9];
  const float* Wv   = (const float*)d_in[10];
  const float* bv   = (const float*)d_in[11];
  const float* Wo   = (const float*)d_in[12];
  const float* bo   = (const float*)d_in[13];
  const float* gq   = (const float*)d_in[14];
  const float* gk   = (const float*)d_in[15];
  const float* mod  = (const float*)d_in[16];
  const float* Wh   = (const float*)d_in[17];
  const float* bh   = (const float*)d_in[18];
  float* out = (float*)d_out;

  char* wsb = (char*)d_ws;
  size_t off = 0;
  auto take = [&](size_t bytes) -> void* {
    void* p = wsb + off;
    off += (bytes + 255) & ~(size_t)255;
    return p;
  };
  unsigned short* wqb  = (unsigned short*)take((size_t)C_N * C_N * 2);  // 4.7 MB
  unsigned short* wob  = (unsigned short*)take((size_t)C_N * C_N * 2);  // 4.7 MB
  unsigned short* qhb  = (unsigned short*)take((size_t)M_N * C_N * 2);  // 100.6 MB (bf16 q_hidden, L2-resident)
  unsigned short* qbf  = (unsigned short*)take((size_t)M_N * C_N * 2);  // 100.6 MB (q roped; reused as o*heat)
  float*          heat = (float*)take((size_t)M_N * 4);
  float*          kws  = (float*)take((size_t)T_N * S_N * C_N * 4);
  float*          vws  = (float*)take((size_t)T_N * S_N * C_N * 4);
  float*          rsq  = (float*)take((size_t)M_N * 4);

  zero_f32<<<(M_N + 255) / 256, 256, 0, stream>>>(rsq, M_N);
  cvt_w<<<(C_N * C_N) / (256 * 4), 256, 0, stream>>>(Wq, Wo, wqb, wob);
  heat_kernel<<<M_N, 256, 0, stream>>>(qh, te, mod, Wh, bh, heat, qhb);
  gemm_wmma<1><<<dim3((M_N + 63) / 64, NH_N), 256, 0, stream>>>(
      qhb, wqb, bq, gq, qf, rsq, qbf, nullptr);
  kv_kernel<<<T_N * S_N, 256, 0, stream>>>(kvh, Wk, bk, Wv, bv, gk, kvf, kws, vws);
  attn_kernel<<<dim3(NH_N, T_N), 256, 0, stream>>>(qbf, kws, vws, heat, rsq, qbf);
  gemm_wmma<0><<<dim3((M_N + 63) / 64, NH_N), 256, 0, stream>>>(
      qbf, wob, bo, nullptr, nullptr, nullptr, nullptr, out);
}
